// GNN_skip_small_83932250898483
// MI455X (gfx1250) — compile-verified, matrix operand order checked by reading, not run on hardware
//
#include <hip/hip_runtime.h>
#include <hip/hip_bf16.h>
#include <math.h>

// Problem constants (from reference): N=16384, F=H1=H2=128, G=64
#define NN 16384
#define FD 128
#define GSEG 64

typedef __attribute__((ext_vector_type(16))) __bf16 v16bf;
typedef __attribute__((ext_vector_type(8)))  float  v8f;

struct alignas(16) U4 { unsigned u[4]; };

union FragAB {
  v16bf v;
  unsigned short s[16];
  U4 q[2];
};

__device__ __forceinline__ unsigned short f2bf(float f) {
  union { float f; unsigned u; } x; x.f = f;
  unsigned r = x.u + 0x7FFFu + ((x.u >> 16) & 1u);   // round-to-nearest-even
  return (unsigned short)(r >> 16);
}

// ---------------------------------------------------------------------------
// C[NN x 128] = A[NN x NN](fp32, row-major) @ B, where B is provided as
// BT[col][k] in bf16 (column-major of the 128-wide matrix). bf16 WMMA.
// Block = 256 threads = 8 waves; block tile 128x128; wave tile 32x64.
// ---------------------------------------------------------------------------
__global__ __launch_bounds__(256) void gemm_bf16(const float* __restrict__ A,
                                                 const unsigned short* __restrict__ BT,
                                                 float* __restrict__ C) {
  const int lane    = threadIdx.x & 31;
  const int w       = threadIdx.x >> 5;
  const int rowWave = blockIdx.x * 128 + (w & 3) * 32;   // 32 rows per wave
  const int colWave = (w >> 2) * 64;                     // 64 cols per wave
  const int l16     = lane & 15;
  const int kb      = (lane < 16) ? 0 : 8;               // ISA 16-bit A/B k-base per half-wave

  v8f zero = {0.f,0.f,0.f,0.f,0.f,0.f,0.f,0.f};
  v8f acc[2][4];
  #pragma unroll
  for (int i = 0; i < 2; ++i)
    #pragma unroll
    for (int j = 0; j < 4; ++j) acc[i][j] = zero;

  const float* arow[2];
  arow[0] = A + (size_t)(rowWave + l16) * NN + kb;
  arow[1] = arow[0] + (size_t)16 * NN;
  const unsigned short* bcol[4];
  #pragma unroll
  for (int j = 0; j < 4; ++j)
    bcol[j] = BT + (size_t)(colWave + j * 16 + l16) * NN + kb;

  #pragma unroll 1
  for (int k0 = 0; k0 < NN; k0 += 32) {
    FragAB Af[2], Bf[4];
    #pragma unroll
    for (int i = 0; i < 2; ++i) {
      const float* p = arow[i] + k0;
      float4 a0 = *(const float4*)(p);
      float4 a1 = *(const float4*)(p + 4);
      float4 a2 = *(const float4*)(p + 16);   // K = kb+16 .. kb+23
      float4 a3 = *(const float4*)(p + 20);
      Af[i].s[0]  = f2bf(a0.x); Af[i].s[1]  = f2bf(a0.y);
      Af[i].s[2]  = f2bf(a0.z); Af[i].s[3]  = f2bf(a0.w);
      Af[i].s[4]  = f2bf(a1.x); Af[i].s[5]  = f2bf(a1.y);
      Af[i].s[6]  = f2bf(a1.z); Af[i].s[7]  = f2bf(a1.w);
      Af[i].s[8]  = f2bf(a2.x); Af[i].s[9]  = f2bf(a2.y);
      Af[i].s[10] = f2bf(a2.z); Af[i].s[11] = f2bf(a2.w);
      Af[i].s[12] = f2bf(a3.x); Af[i].s[13] = f2bf(a3.y);
      Af[i].s[14] = f2bf(a3.z); Af[i].s[15] = f2bf(a3.w);
    }
    #pragma unroll
    for (int j = 0; j < 4; ++j) {
      Bf[j].q[0] = *(const U4*)(bcol[j] + k0);
      Bf[j].q[1] = *(const U4*)(bcol[j] + k0 + 16);
    }
    // prefetch next k-step of adj (global_prefetch_b8)
    if (k0 + 32 < NN) __builtin_prefetch(arow[0] + k0 + 32, 0, 0);

    #pragma unroll
    for (int i = 0; i < 2; ++i)
      #pragma unroll
      for (int j = 0; j < 4; ++j)
        acc[i][j] = __builtin_amdgcn_wmma_f32_16x16x32_bf16(
            false, Af[i].v, false, Bf[j].v, (short)0, acc[i][j], false, false);
  }

  // C/D layout: VGPR r -> M=r (lanes 0-15), M=8+r (lanes 16-31); N = lane%16
  const int rhalf = (lane >> 4) * 8;
  #pragma unroll
  for (int i = 0; i < 2; ++i)
    #pragma unroll
    for (int j = 0; j < 4; ++j) {
      const int rb = rowWave + i * 16 + rhalf;
      const int cb = colWave + j * 16 + l16;
      #pragma unroll
      for (int r = 0; r < 8; ++r)
        C[(size_t)(rb + r) * FD + cb] = acc[i][j][r];
    }
}

// XT[c][k] = bf16(X[k][c])   (X: NN x 128 fp32 row-major)
__global__ void transpose_cvt(const float* __restrict__ X, unsigned short* __restrict__ XT) {
  long long i = (long long)blockIdx.x * blockDim.x + threadIdx.x;
  if (i >= (long long)NN * FD) return;
  int k = (int)(i % NN);
  int c = (int)(i / NN);
  XT[(size_t)c * NN + k] = f2bf(X[(size_t)k * FD + c]);
}

// xpre = relu([Xa | Xb] @ Wm + bias); accumulate per-column sum / sumsq
__global__ __launch_bounds__(128) void mlp_layer(const float* __restrict__ Xa,
                                                 const float* __restrict__ Xb,
                                                 const float* __restrict__ Wm,
                                                 const float* __restrict__ bias,
                                                 float* __restrict__ xpre,
                                                 float* __restrict__ colsum,
                                                 float* __restrict__ colsq) {
  __shared__ float Wl[64][128];
  __shared__ float xl[32][64];
  const int c    = threadIdx.x;
  const int row0 = blockIdx.x * 32;

  float acc[32];
  #pragma unroll
  for (int r = 0; r < 32; ++r) acc[r] = 0.f;

  for (int cc = 0; cc < 4; ++cc) {
    #pragma unroll 4
    for (int t = 0; t < 64; ++t)
      Wl[t][c] = Wm[(size_t)(cc * 64 + t) * 128 + c];
    const float* src = (cc < 2) ? Xa : Xb;
    const int scol = (cc & 1) * 64;
    #pragma unroll
    for (int u = 0; u < 16; ++u) {
      int e = c + u * 128;
      int r = e >> 6, kk = e & 63;
      xl[r][kk] = src[(size_t)(row0 + r) * 128 + scol + kk];
    }
    __syncthreads();
    for (int r = 0; r < 32; ++r) {
      float s = acc[r];
      #pragma unroll 8
      for (int kk = 0; kk < 64; ++kk) s += xl[r][kk] * Wl[kk][c];
      acc[r] = s;
    }
    __syncthreads();
  }

  float s = 0.f, sq = 0.f;
  const float b = bias[c];
  for (int r = 0; r < 32; ++r) {
    float v = fmaxf(acc[r] + b, 0.f);
    xpre[(size_t)(row0 + r) * 128 + c] = v;
    s += v; sq += v * v;
  }
  atomicAdd(&colsum[c], s);
  atomicAdd(&colsq[c], sq);
}

__global__ void bn_finalize(const float* __restrict__ sum, const float* __restrict__ sq,
                            float* __restrict__ mu, float* __restrict__ rs) {
  int c = threadIdx.x;
  float m = sum[c] * (1.f / (float)NN);
  float v = sq[c] * (1.f / (float)NN) - m * m;
  mu[c] = m;
  rs[c] = rsqrtf(v + 1e-5f);
}

// x = where(mask, 0, (xpre-mu)*rs*g + be); optionally infset[i] = (rowmean>0)?0:1
__global__ __launch_bounds__(128) void bn_apply(const float* __restrict__ xpre,
                                                const float* __restrict__ mu,
                                                const float* __restrict__ rs,
                                                const float* __restrict__ g,
                                                const float* __restrict__ be,
                                                const int* __restrict__ mask,
                                                float* __restrict__ xo,
                                                float* __restrict__ infset) {
  __shared__ float red[128];
  const int i = blockIdx.x, c = threadIdx.x;
  float t = (xpre[(size_t)i * 128 + c] - mu[c]) * rs[c] * g[c] + be[c];
  float v = mask[i] ? 0.f : t;
  xo[(size_t)i * 128 + c] = v;
  if (infset) {
    red[c] = v; __syncthreads();
    #pragma unroll
    for (int off = 64; off > 0; off >>= 1) {
      if (c < off) red[c] += red[c + off];
      __syncthreads();
    }
    if (c == 0) {
      float h = red[0] * (1.f / 128.f);
      infset[i] = (h > 0.f) ? 0.f : 1.f;   // sign(h): >0 -> 0, <=0 -> 1
    }
  }
}

__global__ void segment_accum(const float* __restrict__ x_in, const float* __restrict__ x1,
                              const float* __restrict__ x2, const int* __restrict__ idx,
                              float* __restrict__ hg) {
  long long e = (long long)blockIdx.x * 256 + threadIdx.x;
  if (e >= (long long)NN * 384) return;
  int i = (int)(e / 384), c = (int)(e % 384);
  float v;
  if (c < 128)      v = x_in[(size_t)i * 128 + c];
  else if (c < 256) v = x1[(size_t)i * 128 + (c - 128)];
  else              v = x2[(size_t)i * 128 + (c - 256)];
  atomicAdd(&hg[(size_t)idx[i] * 384 + c], v);
}

__global__ __launch_bounds__(128) void out_head(const float* __restrict__ hg,
                                                const float* __restrict__ W4,
                                                const float* __restrict__ b4,
                                                float* __restrict__ out) {
  __shared__ float red[128];
  const int gseg = blockIdx.x, c = threadIdx.x;
  float s = 0.f;
  for (int u = c; u < 384; u += 128) s += hg[(size_t)gseg * 384 + u] * W4[u];
  red[c] = s; __syncthreads();
  #pragma unroll
  for (int off = 64; off > 0; off >>= 1) {
    if (c < off) red[c] += red[c + off];
    __syncthreads();
  }
  if (c == 0) out[gseg] = fmaxf(red[0] + b4[0], 0.f);
}

__global__ __launch_bounds__(256) void ovr_matvec(const float* __restrict__ pos,
                                                  const float* __restrict__ infset,
                                                  float* __restrict__ ovr) {
  __shared__ float red[256];
  const int i = blockIdx.x, t = threadIdx.x;
  const float4* p = (const float4*)(pos + (size_t)i * NN);
  const float4* f = (const float4*)infset;
  float s = 0.f;
  for (int u = t; u < NN / 4; u += 256) {
    float4 a = p[u]; float4 b = f[u];
    s += a.x * b.x + a.y * b.y + a.z * b.z + a.w * b.w;
  }
  red[t] = s; __syncthreads();
  #pragma unroll
  for (int off = 128; off > 0; off >>= 1) {
    if (t < off) red[t] += red[t + off];
    __syncthreads();
  }
  if (t == 0) ovr[i] = red[0];
}

__global__ void seed_mask(const int* __restrict__ seed, float* __restrict__ ovr) {
  int t = threadIdx.x;
  if (t < 10) ovr[seed[t]] = -INFINITY;
}

__global__ __launch_bounds__(256) void argmax_stage1(const float* __restrict__ ovr,
                                                     float* __restrict__ pmax,
                                                     int* __restrict__ pidx) {
  __shared__ float rv[256];
  __shared__ int   ri[256];
  const int t = threadIdx.x;
  const int i = blockIdx.x * 256 + t;
  rv[t] = ovr[i]; ri[t] = i; __syncthreads();
  #pragma unroll
  for (int off = 128; off > 0; off >>= 1) {
    if (t < off) {
      if (rv[t + off] > rv[t] || (rv[t + off] == rv[t] && ri[t + off] < ri[t])) {
        rv[t] = rv[t + off]; ri[t] = ri[t + off];
      }
    }
    __syncthreads();
  }
  if (t == 0) { pmax[blockIdx.x] = rv[0]; pidx[blockIdx.x] = ri[0]; }
}

__global__ void argmax_stage2(const float* __restrict__ pmax, const int* __restrict__ pidx,
                              float* __restrict__ slot) {
  __shared__ float rv[64];
  __shared__ int   ri[64];
  const int t = threadIdx.x;
  rv[t] = pmax[t]; ri[t] = pidx[t]; __syncthreads();
  #pragma unroll
  for (int off = 32; off > 0; off >>= 1) {
    if (t < off) {
      if (rv[t + off] > rv[t] || (rv[t + off] == rv[t] && ri[t + off] < ri[t])) {
        rv[t] = rv[t + off]; ri[t] = ri[t + off];
      }
    }
    __syncthreads();
  }
  if (t == 0) *slot = (float)ri[0];
}

__global__ void zero_f32(float* __restrict__ p, int n) {
  int i = blockIdx.x * 256 + threadIdx.x;
  if (i < n) p[i] = 0.f;
}

// ---------------------------------------------------------------------------
// Workspace layout (float elements): requires ~38 MB of d_ws.
//   [0,        1048576)  XT (bf16, 128 x 16384, viewed as ushort)
//   [1048576,  3145728)  ax   (NN x 128)
//   [3145728,  5242880)  xpre (NN x 128)
//   [5242880,  7340032)  x1   (NN x 128)
//   [7340032,  9437184)  x2   (NN x 128)
//   [9437184,  9438208)  stats: sum1,sq1,mu1,rs1,sum2,sq2,mu2,rs2 (8 x 128)
//   [9438208,  9462784)  hg (64 x 384)
//   [9462784,  9479168)  ovr (NN)
//   [9479168,  9479232)  pmax (64)  then pidx (64 ints)
// d_out: [0,64) out; [64,64+NN) infset; [64+NN] new_seed (as float)
// ---------------------------------------------------------------------------
extern "C" void kernel_launch(void* const* d_in, const int* in_sizes, int n_in,
                              void* d_out, int out_size, void* d_ws, size_t ws_size,
                              hipStream_t stream) {
  const float* adj  = (const float*)d_in[0];
  const float* x_in = (const float*)d_in[1];
  const int*   mask = (const int*)d_in[2];
  const float* pos  = (const float*)d_in[3];
  const int*   idx  = (const int*)d_in[4];
  const int*   seed = (const int*)d_in[5];
  // d_in[6] = step (unused by reference)
  const float* W1  = (const float*)d_in[7];
  const float* b1  = (const float*)d_in[8];
  const float* W2  = (const float*)d_in[9];
  const float* b2  = (const float*)d_in[10];
  const float* W4  = (const float*)d_in[11];
  const float* b4  = (const float*)d_in[12];
  const float* g1  = (const float*)d_in[13];
  const float* be1 = (const float*)d_in[14];
  const float* g2  = (const float*)d_in[15];
  const float* be2 = (const float*)d_in[16];

  float* W = (float*)d_ws;
  unsigned short* XT = (unsigned short*)d_ws;
  float* ax    = W + 1048576;
  float* xpre  = W + 3145728;
  float* x1    = W + 5242880;
  float* x2    = W + 7340032;
  float* stats = W + 9437184;
  float* hg    = stats + 1024;
  float* ovr   = hg + 24576;
  float* pmax  = ovr + NN;
  int*   pidx  = (int*)(pmax + 64);

  float* outv     = (float*)d_out;        // [0,64)
  float* infset   = outv + 64;            // [64, 64+NN)
  float* seedslot = outv + 64 + NN;       // last element

  // zero stats + hg (contiguous 25600 floats)
  zero_f32<<<(25600 + 255) / 256, 256, 0, stream>>>(stats, 25600);

  // layer 1: ax = adj @ x_in (bf16 WMMA)
  transpose_cvt<<<(NN * FD) / 256, 256, 0, stream>>>(x_in, XT);
  gemm_bf16<<<NN / 128, 256, 0, stream>>>(adj, XT, ax);
  mlp_layer<<<NN / 32, 128, 0, stream>>>(x_in, ax, W1, b1, xpre, stats, stats + 128);
  bn_finalize<<<1, 128, 0, stream>>>(stats, stats + 128, stats + 256, stats + 384);
  bn_apply<<<NN, 128, 0, stream>>>(xpre, stats + 256, stats + 384, g1, be1, mask, x1, infset);

  // layer 2: ax = adj @ x1
  transpose_cvt<<<(NN * FD) / 256, 256, 0, stream>>>(x1, XT);
  gemm_bf16<<<NN / 128, 256, 0, stream>>>(adj, XT, ax);
  mlp_layer<<<NN / 32, 128, 0, stream>>>(x1, ax, W2, b2, xpre, stats + 512, stats + 640);
  bn_finalize<<<1, 128, 0, stream>>>(stats + 512, stats + 640, stats + 768, stats + 896);
  bn_apply<<<NN, 128, 0, stream>>>(xpre, stats + 768, stats + 896, g2, be2, mask, x2, nullptr);

  // segment sum + head
  segment_accum<<<(NN * 384) / 256, 256, 0, stream>>>(x_in, x1, x2, idx, hg);
  out_head<<<GSEG, 128, 0, stream>>>(hg, W4, b4, outv);

  // ovr = pos @ infset; mask seeds; argmax
  ovr_matvec<<<NN, 256, 0, stream>>>(pos, infset, ovr);
  seed_mask<<<1, 32, 0, stream>>>(seed, ovr);
  argmax_stage1<<<NN / 256, 256, 0, stream>>>(ovr, pmax, pidx);
  argmax_stage2<<<1, 64, 0, stream>>>(pmax, pidx, seedslot);
}
